// M2MGNN_14370960572528
// MI455X (gfx1250) — matrix operand order repaired and checked by reading, not compiled, not run on hardware
//
#include <hip/hip_runtime.h>
#include <hip/hip_bf16.h>

#define GN      50000
#define GIN_C   256
#define GHID    64
#define GC      4
#define GHC     256      // HID * C
#define GOUT_C  64
#define GE      800000
#define GL      2
#define GBETA   0.1f
#define GEPS    1e-5f

typedef __attribute__((ext_vector_type(16))) _Float16 v16h;
typedef __attribute__((ext_vector_type(8)))  float    v8f;

// ---------------------------------------------------------------------------
// Convert input activations x to f16 (one-time, GEMM A-side feed).
// ---------------------------------------------------------------------------
__global__ void cvt_x_kernel(const float* __restrict__ x, _Float16* __restrict__ xh) {
    const size_t i = (size_t)blockIdx.x * blockDim.x + threadIdx.x;
    xh[i] = (_Float16)x[i];
}

// ---------------------------------------------------------------------------
// Swizzle a weight matrix [256, Ncols] (f32) into WMMA B-fragment order (f16):
// for column-tile nt, k-step ks, lane l: 16 contiguous halves =
//   B[(ks*32 + 16*(l>=16) + j) * Ncols + nt*16 + (l&15)],  j = 0..15
// One wave per (nt, ks). grid = Ncols/16, block = 256 (8 waves = 8 k-steps).
// ---------------------------------------------------------------------------
__global__ void swz_weight_kernel(const float* __restrict__ src,
                                  _Float16* __restrict__ dst, int Ncols) {
    const int nt = blockIdx.x;
    const int ks = threadIdx.x >> 5;
    const int l  = threadIdx.x & 31;
    const int kh = (l >> 4) << 4;
    const int cn = l & 15;
    v16h t;
#pragma unroll
    for (int j = 0; j < 16; ++j)
        t[j] = (_Float16)src[(size_t)(ks * 32 + kh + j) * Ncols + nt * 16 + cn];
    *(v16h*)(dst + (((size_t)nt * 8 + ks) * 32 + l) * 16) = t;
}

// ---------------------------------------------------------------------------
// WMMA GEMM: out[M,Ncols](f32) = Ah[M,256](f16) @ Bswz(f16, fragment-major)
//            (+bias) (relu?)
// grid.x = M/16 ; blockDim.x = (Ncols/16)*32 (one wave per 16x16 tile).
// The 16x256 f16 A strip (8KB) is staged to LDS with the CDNA5 async-to-LDS
// path (GLOBAL_LOAD_ASYNC_TO_LDS_B128, ASYNCcnt), then each wave reads its
// A fragments from LDS (ds_load_b128) and B fragments as one v16h global load.
// ---------------------------------------------------------------------------
#define KPADH 264   // halves per LDS row (256 + 8 pad)
__global__ void wmma_gemm256_kernel(const _Float16* __restrict__ Ah,
                                    const _Float16* __restrict__ Bswz,
                                    const float* __restrict__ bias,
                                    float* __restrict__ out,
                                    int Ncols, int do_relu) {
    __shared__ _Float16 As[16 * KPADH];

    const int tid   = threadIdx.x;
    const int lane  = tid & 31;
    const int wave  = tid >> 5;
    const int row0  = blockIdx.x * 16;
    const int ncol0 = wave * 16;

    // Async-stage the A strip: 512 x 16-byte chunks (32 chunks per row, 16 rows).
    for (int c = tid; c < 512; c += blockDim.x) {
        const int m  = c >> 5;           // row within strip
        const int h8 = (c & 31) << 3;    // starting half within row
        const unsigned lds_off = (unsigned)(size_t)&As[m * KPADH + h8];
        const _Float16* gsrc = Ah + (size_t)(row0 + m) * 256 + h8;
        asm volatile("global_load_async_to_lds_b128 %0, %1, off"
                     :: "v"(lds_off), "v"(gsrc) : "memory");
    }
    asm volatile("s_wait_asynccnt 0" ::: "memory");
    __syncthreads();

    const int m  = lane & 15;            // A row within tile
    const int kb = (lane >> 4) << 4;     // A: 0 or 16
    const int cn = lane & 15;            // B/D column within tile

    const v16h* Bf = (const v16h*)Bswz + (size_t)wave * 256 + lane;

    v8f acc = {};
#pragma unroll
    for (int k0 = 0; k0 < 256; k0 += 32) {
        v16h av;
#pragma unroll
        for (int j = 0; j < 8; ++j) {
            av[j]     = As[m * KPADH + k0 + kb + j];        // K = kb + j
            av[8 + j] = As[m * KPADH + k0 + kb + 8 + j];    // K = kb + 8 + j
        }
        const v16h bv = Bf[(k0 >> 5) * 32];
        acc = __builtin_amdgcn_wmma_f32_16x16x32_f16(
            /*neg_a=*/false, av, /*neg_b=*/false, bv,
            /*c_mod=*/(short)0, acc, /*reuse_a=*/false, /*reuse_b=*/false);
    }

    const float bv_ = bias ? bias[ncol0 + cn] : 0.0f;
    const int mbase = row0 + ((lane >> 4) << 3);   // M = r + 8*(lane>=16)
#pragma unroll
    for (int r = 0; r < 8; ++r) {
        float v = acc[r] + bv_;
        if (do_relu) v = fmaxf(v, 0.0f);
        out[(size_t)(mbase + r) * Ncols + ncol0 + cn] = v;
    }
}

// ---------------------------------------------------------------------------
// LayerNorm over HC=256 per node row; also maintains the f16 shadow of x.
// mode 0: x = LN(x)*g+b ; ego = x ; xh = f16(x)
// mode 1: x = 0.9*(LN(relu(x))*g+b) + 0.1*ego ; xh = f16(x)
// blockDim = 256, grid = N
// ---------------------------------------------------------------------------
__global__ void ln_kernel(float* __restrict__ x, float* __restrict__ ego,
                          _Float16* __restrict__ xh,
                          const float* __restrict__ g, const float* __restrict__ b,
                          int mode) {
    __shared__ float s1[256];
    __shared__ float s2[256];
    const int rowi = blockIdx.x;
    const int t    = threadIdx.x;
    const size_t idx = (size_t)rowi * GHC + t;

    float v = x[idx];
    if (mode == 1) v = fmaxf(v, 0.0f);
    s1[t] = v;
    s2[t] = v * v;
    __syncthreads();
#pragma unroll
    for (int off = 128; off > 0; off >>= 1) {
        if (t < off) { s1[t] += s1[t + off]; s2[t] += s2[t + off]; }
        __syncthreads();
    }
    const float mu  = s1[0] * (1.0f / 256.0f);
    const float var = s2[0] * (1.0f / 256.0f) - mu * mu;
    float y = (v - mu) * rsqrtf(var + GEPS) * g[t] + b[t];
    if (mode == 0) {
        ego[idx] = y;
    } else {
        y = (1.0f - GBETA) * y + GBETA * ego[idx];
    }
    x[idx]  = y;
    xh[idx] = (_Float16)y;
}

// ---------------------------------------------------------------------------
// Edge kernel: one wave32 per edge. Lane owns dims {2*lane, 2*lane+1} of HID.
//   t = relu(0.5*h[row] + h[col]); br = t @ att  (4 logits, wave-reduced)
//   a = softmax(br); acc[row][k*64+d] += a[k]*h[col][d]   (256 atomics/edge)
// blockDim = 256 (8 edges/block), grid = E/8
// ---------------------------------------------------------------------------
__global__ void edge_kernel(const int* __restrict__ row, const int* __restrict__ col,
                            const float* __restrict__ h,
                            const float* __restrict__ att,   // [64,4]
                            float* __restrict__ acc) {
    const int e = blockIdx.x * (blockDim.x >> 5) + (threadIdx.x >> 5);
    if (e >= GE) return;
    const int lane = threadIdx.x & 31;
    const int d0 = lane << 1;

    const int r = row[e];
    const int c = col[e];

    // Hint the accumulator row toward the caches before the atomic burst.
    __builtin_prefetch(&acc[(size_t)r * GHC + d0], 1, 3);

    const float2 hr = ((const float2*)(h + (size_t)r * GHID))[lane];
    const float2 hc = ((const float2*)(h + (size_t)c * GHID))[lane];

    const float t0 = fmaxf(0.5f * hr.x + hc.x, 0.0f);
    const float t1 = fmaxf(0.5f * hr.y + hc.y, 0.0f);

    const float4 a0 = ((const float4*)att)[d0];
    const float4 a1 = ((const float4*)att)[d0 + 1];

    float br[4];
    br[0] = t0 * a0.x + t1 * a1.x;
    br[1] = t0 * a0.y + t1 * a1.y;
    br[2] = t0 * a0.z + t1 * a1.z;
    br[3] = t0 * a0.w + t1 * a1.w;
#pragma unroll
    for (int off = 16; off >= 1; off >>= 1) {
#pragma unroll
        for (int k = 0; k < 4; ++k) br[k] += __shfl_xor(br[k], off, 32);
    }
    const float mx = fmaxf(fmaxf(br[0], br[1]), fmaxf(br[2], br[3]));
    float a[4], s = 0.0f;
#pragma unroll
    for (int k = 0; k < 4; ++k) { a[k] = __expf(br[k] - mx); s += a[k]; }
    const float inv = 1.0f / s;

    float* dst = &acc[(size_t)r * GHC];
#pragma unroll
    for (int k = 0; k < 4; ++k) {
        const float w = a[k] * inv;
        atomicAdd(&dst[k * GHID + d0],     w * hc.x);
        atomicAdd(&dst[k * GHID + d0 + 1], w * hc.y);
    }
}

// ---------------------------------------------------------------------------
// log_softmax over OUT_C=64 per row. One wave per row, 8 rows per block.
// ---------------------------------------------------------------------------
__global__ void logsoftmax64_kernel(const float* __restrict__ logits,
                                    float* __restrict__ out) {
    const int rowi = blockIdx.x * (blockDim.x >> 5) + (threadIdx.x >> 5);
    if (rowi >= GN) return;
    const int lane = threadIdx.x & 31;

    const float v0 = logits[(size_t)rowi * GOUT_C + lane];
    const float v1 = logits[(size_t)rowi * GOUT_C + 32 + lane];

    float m = fmaxf(v0, v1);
#pragma unroll
    for (int off = 16; off >= 1; off >>= 1) m = fmaxf(m, __shfl_xor(m, off, 32));
    float s = __expf(v0 - m) + __expf(v1 - m);
#pragma unroll
    for (int off = 16; off >= 1; off >>= 1) s += __shfl_xor(s, off, 32);
    const float lse = m + __logf(s);

    out[(size_t)rowi * GOUT_C + lane]      = v0 - lse;
    out[(size_t)rowi * GOUT_C + 32 + lane] = v1 - lse;
}

// ---------------------------------------------------------------------------
extern "C" void kernel_launch(void* const* d_in, const int* in_sizes, int n_in,
                              void* d_out, int out_size, void* d_ws, size_t ws_size,
                              hipStream_t stream) {
    const float* x    = (const float*)d_in[0];
    const int*   ei   = (const int*)  d_in[1];   // [2,E]
    const float* W1   = (const float*)d_in[2];
    const float* b1   = (const float*)d_in[3];
    const float* lin  = (const float*)d_in[4];   // [L,HC,HID]
    const float* att  = (const float*)d_in[5];   // [L,HID,C]
    const float* lng  = (const float*)d_in[6];   // [L+1,HC]
    const float* lnb  = (const float*)d_in[7];
    const float* W2   = (const float*)d_in[8];
    const float* b2   = (const float*)d_in[9];
    float* out = (float*)d_out;

    // Workspace carve-up (all offsets 256B-aligned).
    char* p = (char*)d_ws;
    float*    xb   = (float*)p;    p += (size_t)GN * GHC * sizeof(float);     // 51.2 MB
    float*    ego  = (float*)p;    p += (size_t)GN * GHC * sizeof(float);     // 51.2 MB
    float*    hb   = (float*)p;    p += (size_t)GN * GHID * sizeof(float);    // 12.8 MB
    _Float16* xh   = (_Float16*)p; p += (size_t)GN * GHC * sizeof(_Float16);  // 25.6 MB
    _Float16* W1s  = (_Float16*)p; p += (size_t)GIN_C * GHC * sizeof(_Float16);
    _Float16* lins = (_Float16*)p; p += (size_t)GL * GHC * GHID * sizeof(_Float16);
    _Float16* W2s  = (_Float16*)p; p += (size_t)GHC * GOUT_C * sizeof(_Float16);

    // 0) f16 input copy + fragment-swizzled f16 weights
    cvt_x_kernel<<<(GN * GHC) / 256, 256, 0, stream>>>(x, xh);
    swz_weight_kernel<<<GHC / 16,   256, 0, stream>>>(W1, W1s, GHC);
    swz_weight_kernel<<<GHID / 16,  256, 0, stream>>>(lin,                 lins,                 GHID);
    swz_weight_kernel<<<GHID / 16,  256, 0, stream>>>(lin + GHC * GHID,    lins + GHC * GHID,    GHID);
    swz_weight_kernel<<<GOUT_C / 16, 256, 0, stream>>>(W2, W2s, GOUT_C);

    // 1) x = relu(x @ W1 + b1)   [N,256] via WMMA
    wmma_gemm256_kernel<<<GN / 16, (GHC / 16) * 32, 0, stream>>>(
        xh, W1s, b1, xb, GHC, /*relu=*/1);

    // 2) x = LN(x); ego = x; xh = f16(x)
    ln_kernel<<<GN, 256, 0, stream>>>(xb, ego, xh, lng, lnb, /*mode=*/0);

    for (int i = 0; i < GL; ++i) {
        // 3) h = x @ lin_ws[i]   [N,64] via WMMA
        wmma_gemm256_kernel<<<GN / 16, (GHID / 16) * 32, 0, stream>>>(
            xh, lins + (size_t)i * GHC * GHID / 2 * 2, nullptr, hb, GHID, /*relu=*/0);

        // 4) zero accumulator (f32 x is dead once h is computed)
        hipMemsetAsync(xb, 0, (size_t)GN * GHC * sizeof(float), stream);

        // 5) edge gather / gated softmax / scatter-add
        edge_kernel<<<GE / 8, 256, 0, stream>>>(
            ei, ei + GE, hb, att + (size_t)i * GHID * GC, xb);

        // 6) x = 0.9*LN(relu(acc)) + 0.1*ego ; xh = f16(x)
        ln_kernel<<<GN, 256, 0, stream>>>(
            xb, ego, xh, lng + (size_t)(i + 1) * GHC, lnb + (size_t)(i + 1) * GHC,
            /*mode=*/1);
    }

    // 7) logits = x @ W2 + b2   [N,64] via WMMA (reuse h buffer)
    wmma_gemm256_kernel<<<GN / 16, (GOUT_C / 16) * 32, 0, stream>>>(
        xh, W2s, b2, hb, GOUT_C, /*relu=*/0);

    // 8) log_softmax -> d_out
    logsoftmax64_kernel<<<(GN + 7) / 8, 256, 0, stream>>>(hb, out);
}
